// ContinuousMambaBlock_71236327571715
// MI455X (gfx1250) — matrix operand order repaired and checked by reading, not compiled, not run on hardware
//
#include <hip/hip_runtime.h>

// ---------------------------------------------------------------------------
// ContinuousMambaBlock fused pipeline for gfx1250 (MI455X), wave32 + WMMA f16
// Shapes: B=512, DM=1024, DI=2048, DS=64, DTR=64, DT=64, G=8, E=4, K=2, DC=4
// Each wave computes a 32x64 output tile: 2 A fragments share 4 B fragments,
// 8 WMMAs per 32-K chunk, raw f32 loads double-buffered one chunk ahead.
// ---------------------------------------------------------------------------

typedef __attribute__((ext_vector_type(16))) _Float16 v16h;
typedef __attribute__((ext_vector_type(8)))  float    v8f;

__device__ inline float4 ldg4(const float* p) {
    return *reinterpret_cast<const float4*>(p);
}

__device__ inline void cvt4(v16h& dst, int base, float4 f) {
    dst[base + 0] = (_Float16)f.x;
    dst[base + 1] = (_Float16)f.y;
    dst[base + 2] = (_Float16)f.z;
    dst[base + 3] = (_Float16)f.w;
}

__device__ inline float siluf(float v)     { return v / (1.0f + expf(-v)); }
__device__ inline float softplusf(float v) { return fmaxf(v, 0.0f) + log1pf(expf(-fabsf(v))); }

// Raw (f32) chunk: two 16x32 A fragment feeds + 4x 32x16 B feeds (24 float4).
struct Raw {
    float4 a[2][4];
    float4 b[4][4];
};

__device__ inline void load_chunk(Raw& r,
                                  const float* __restrict__ Ar0,
                                  const float* __restrict__ Ar1,
                                  const float* __restrict__ Br0,
                                  const float* __restrict__ Br1,
                                  const float* __restrict__ Br2,
                                  const float* __restrict__ Br3,
                                  int kA, int kB) {
    // Issue all 24 b128 loads back-to-back; consumed only in compute_chunk.
    r.a[0][0] = ldg4(Ar0 + kA);
    r.a[0][1] = ldg4(Ar0 + kA + 4);
    r.a[0][2] = ldg4(Ar0 + kA + 16);
    r.a[0][3] = ldg4(Ar0 + kA + 20);
    r.a[1][0] = ldg4(Ar1 + kA);
    r.a[1][1] = ldg4(Ar1 + kA + 4);
    r.a[1][2] = ldg4(Ar1 + kA + 16);
    r.a[1][3] = ldg4(Ar1 + kA + 20);
    const float* bp[4] = {Br0, Br1, Br2, Br3};
#pragma unroll
    for (int s = 0; s < 4; ++s) {
        r.b[s][0] = ldg4(bp[s] + kB);
        r.b[s][1] = ldg4(bp[s] + kB + 4);
        r.b[s][2] = ldg4(bp[s] + kB + 16);
        r.b[s][3] = ldg4(bp[s] + kB + 20);
    }
}

__device__ inline void compute_chunk(v8f (&acc)[2][4], const Raw& c) {
    v16h af0, af1;
    cvt4(af0, 0,  c.a[0][0]);
    cvt4(af0, 4,  c.a[0][1]);
    cvt4(af0, 8,  c.a[0][2]);
    cvt4(af0, 12, c.a[0][3]);
    cvt4(af1, 0,  c.a[1][0]);
    cvt4(af1, 4,  c.a[1][1]);
    cvt4(af1, 8,  c.a[1][2]);
    cvt4(af1, 12, c.a[1][3]);
#pragma unroll
    for (int s = 0; s < 4; ++s) {
        v16h bf;
        cvt4(bf, 0,  c.b[s][0]);
        cvt4(bf, 4,  c.b[s][1]);
        cvt4(bf, 8,  c.b[s][2]);
        cvt4(bf, 12, c.b[s][3]);
        acc[0][s] = __builtin_amdgcn_wmma_f32_16x16x32_f16(
            false, af0, false, bf, (short)0, acc[0][s], false, false);
        acc[1][s] = __builtin_amdgcn_wmma_f32_16x16x32_f16(
            false, af1, false, bf, (short)0, acc[1][s], false, false);
    }
}

// One uniform-K segment with fixed base pointers, software-pipelined by one
// 32-K chunk (ping-pong raw buffers).  kLen is a positive multiple of 32.
__device__ inline void gemm_segment(v8f (&acc)[2][4],
                                    const float* __restrict__ A, int lda,
                                    const float* __restrict__ B, int ldb,
                                    int kLen, int m0, int lrow, int lhalf, int n0) {
    const float* Ar0 = A + (size_t)(m0 + lrow) * lda;
    const float* Ar1 = A + (size_t)(m0 + 16 + lrow) * lda;
    const float* Br0 = B + (size_t)(n0 + 0 * 16 + lrow) * ldb;
    const float* Br1 = B + (size_t)(n0 + 1 * 16 + lrow) * ldb;
    const float* Br2 = B + (size_t)(n0 + 2 * 16 + lrow) * ldb;
    const float* Br3 = B + (size_t)(n0 + 3 * 16 + lrow) * ldb;
    const int oA = lhalf * 8;   // A: two 8-float runs at kA, kA+16
    const int oB = lhalf * 8;   // B: two 8-float runs at kB, kB+16

    Raw c0, c1;
    load_chunk(c0, Ar0, Ar1, Br0, Br1, Br2, Br3, oA, oB);
    int kk = 0;
    for (; kk + 64 <= kLen; kk += 64) {
        load_chunk(c1, Ar0, Ar1, Br0, Br1, Br2, Br3, kk + 32 + oA, kk + 32 + oB);
        compute_chunk(acc, c0);
        if (kk + 64 < kLen)
            load_chunk(c0, Ar0, Ar1, Br0, Br1, Br2, Br3, kk + 64 + oA, kk + 64 + oB);
        compute_chunk(acc, c1);
    }
    if (kk < kLen)              // odd trailing chunk (already in c0)
        compute_chunk(acc, c0);
}

// ---------------------------------------------------------------------------
// Generic GEMM: out[m][n] = sum_k A[m][k] * W[n][k]  (+ fused epilogue).
// K optionally split at kSplit between (A0,B0) and (A1,B1) sources.
// One wave per 32x64 output tile.
// ---------------------------------------------------------------------------
struct GemmArgs {
    const float* A0; int lda0;
    const float* A1; int lda1;
    const float* B0; int ldb0;
    const float* B1; int ldb1;
    float* out;      int ldc;
    float* out2;               // mode 2: silu(z) destination
    int K; int kSplit;
    int mode;                  // 0=plain 1=+bias 2=conv+silu split 4=ssm gate
    const float* bias;         // mode1: b_in | mode4: dt_bias
    const float* aux0;         // mode2: conv_w (DI,4) | mode4: rowdot (512)
    const float* aux1;         // mode2: conv_b       | mode4: D_skip
    const float* aux2;         // mode4: xact (512x2048)
    const float* aux3;         // mode4: sz   (512x2048)
};

__global__ __launch_bounds__(32) void gemm_f16_wmma(GemmArgs g) {
    const int lane  = threadIdx.x & 31;
    const int lrow  = lane & 15;
    const int lhalf = lane >> 4;
    const int m0    = blockIdx.y * 32;
    const int n0    = blockIdx.x * 64;

    v8f acc[2][4];
#pragma unroll
    for (int h = 0; h < 2; ++h)
#pragma unroll
        for (int s = 0; s < 4; ++s)
#pragma unroll
            for (int r = 0; r < 8; ++r) acc[h][s][r] = 0.0f;

    gemm_segment(acc, g.A0, g.lda0, g.B0, g.ldb0, g.kSplit, m0, lrow, lhalf, n0);
    if (g.K > g.kSplit)
        gemm_segment(acc, g.A1, g.lda1, g.B1, g.ldb1, g.K - g.kSplit,
                     m0, lrow, lhalf, n0);

    // ---- epilogue: lane writes column n, rows m0 + h*16 + lhalf*8 + r ----
#pragma unroll
    for (int h = 0; h < 2; ++h) {
#pragma unroll
        for (int s = 0; s < 4; ++s) {
            const int n = n0 + s * 16 + lrow;
#pragma unroll
            for (int r = 0; r < 8; ++r) {
                const int mr = m0 + h * 16 + lhalf * 8 + r;
                float v = acc[h][s][r];
                if (g.mode == 0) {
                    g.out[(size_t)mr * g.ldc + n] = v;
                } else if (g.mode == 1) {
                    g.out[(size_t)mr * g.ldc + n] = v + g.bias[n];
                } else if (g.mode == 2) {
                    if (n < 2048) {
                        g.out[(size_t)mr * 2048 + n] =
                            siluf(v * g.aux0[n * 4 + 3] + g.aux1[n]);
                    } else {
                        g.out2[(size_t)mr * 2048 + (n - 2048)] = siluf(v);
                    }
                } else { // mode 4: SSM gate
                    float dt = softplusf(v + g.bias[n]);
                    size_t ix = (size_t)mr * 2048 + n;
                    float xa = g.aux2[ix];
                    g.out[ix] = (dt * g.aux0[mr] + g.aux1[n]) * xa * g.aux3[ix];
                }
            }
        }
    }
}

// ---------------------------------------------------------------------------
// Small ops: router softmax + top-2, sin embeddings, KAN RBF difference.
// One 64-thread block per batch row.
// ---------------------------------------------------------------------------
__device__ inline void routerf(float t, const float* rw, const float* rb,
                               float* w, float* msk, float* wn) {
    float lg[4]; float mx = -1e30f;
#pragma unroll
    for (int e = 0; e < 4; ++e) { lg[e] = t * rw[e] + rb[e]; mx = fmaxf(mx, lg[e]); }
    float sum = 0.0f;
#pragma unroll
    for (int e = 0; e < 4; ++e) { w[e] = expf(lg[e] - mx); sum += w[e]; }
#pragma unroll
    for (int e = 0; e < 4; ++e) w[e] /= sum;
    int i1 = 0;
#pragma unroll
    for (int e = 1; e < 4; ++e) if (w[e] > w[i1]) i1 = e;
    int i2 = -1;
#pragma unroll
    for (int e = 0; e < 4; ++e) {
        if (e == i1) continue;
        if (i2 < 0 || w[e] > w[i2]) i2 = e;
    }
#pragma unroll
    for (int e = 0; e < 4; ++e) msk[e] = 0.0f;
    msk[i1] = 1.0f; msk[i2] = 1.0f;
    float ws = w[i1] + w[i2] + 1e-8f;
#pragma unroll
    for (int e = 0; e < 4; ++e) wn[e] = w[e] * msk[e] / ws;
}

__global__ __launch_bounds__(64) void small_ops_kernel(
    const float* __restrict__ tk_cur, const float* __restrict__ tk_prev,
    const float* __restrict__ rW, const float* __restrict__ rb,
    const float* __restrict__ freq, const float* __restrict__ phase,
    const float* __restrict__ kan_W,
    float* __restrict__ dtemb, float* __restrict__ ew, float* __restrict__ em) {
    __shared__ float dphi[64][8];
    const int b = blockIdx.x;
    const int tid = threadIdx.x;

    const float tc = tk_cur[b], tp = tk_prev[b];
    float w_c[4], msk_c[4], wn_c[4], w_p[4], msk_p[4], wn_p[4];
    routerf(tc, rW, rb, w_c, msk_c, wn_c);
    routerf(tp, rW, rb, w_p, msk_p, wn_p);
    if (tid < 4) { ew[b * 4 + tid] = w_c[tid]; em[b * 4 + tid] = msk_c[tid]; }

    float ec = 0.0f, ep = 0.0f;
#pragma unroll
    for (int e = 0; e < 4; ++e) {
        const float f = freq[e * 64 + tid], p = phase[e * 64 + tid];
        ec += wn_c[e] * sinf(tc * f + p);
        ep += wn_p[e] * sinf(tp * f + p);
    }

    const float hh = 2.0f / 7.0f;
#pragma unroll
    for (int gg = 0; gg < 8; ++gg) {
        const float gv = -1.0f + hh * gg;
        const float dc = (ec - gv) / hh;
        const float dp = (ep - gv) / hh;
        dphi[tid][gg] = expf(-dc * dc) - expf(-dp * dp);
    }
    __syncthreads();

    float acc = 0.0f;
    const float* kw = kan_W + (size_t)tid * 64 * 8;
    for (int i = 0; i < 64; ++i)
#pragma unroll
        for (int gg = 0; gg < 8; ++gg)
            acc += dphi[i][gg] * kw[i * 8 + gg];
    dtemb[b * 64 + tid] = acc;
}

// rowdot[b] = sum_s Bmat[b][s] * Cmat[b][s]   (x_dbl cols 64..127 vs 128..191)
__global__ void rowdot_kernel(const float* __restrict__ xdbl,
                              float* __restrict__ rowdot) {
    int b = blockIdx.x * blockDim.x + threadIdx.x;
    if (b >= 512) return;
    const float* r = xdbl + (size_t)b * 192;
    float s = 0.0f;
    for (int i = 0; i < 64; ++i) s += r[64 + i] * r[128 + i];
    rowdot[b] = s;
}

// ---------------------------------------------------------------------------
extern "C" void kernel_launch(void* const* d_in, const int* in_sizes, int n_in,
                              void* d_out, int out_size, void* d_ws, size_t ws_size,
                              hipStream_t stream) {
    const float* uk       = (const float*)d_in[0];
    const float* tkc      = (const float*)d_in[1];
    const float* tkp      = (const float*)d_in[2];
    const float* W_in     = (const float*)d_in[3];
    const float* b_in     = (const float*)d_in[4];
    const float* rW       = (const float*)d_in[5];
    const float* rb       = (const float*)d_in[6];
    const float* freq     = (const float*)d_in[7];
    const float* phase    = (const float*)d_in[8];
    const float* kan_W    = (const float*)d_in[9];
    const float* W_inproj = (const float*)d_in[10];
    const float* conv_w   = (const float*)d_in[11];
    const float* conv_b   = (const float*)d_in[12];
    const float* W_x      = (const float*)d_in[13];
    const float* W_dt     = (const float*)d_in[14];
    const float* dt_bias  = (const float*)d_in[15];
    const float* W_delta  = (const float*)d_in[16];
    // d_in[17] = A_log (unused by reference math)
    const float* D_skip   = (const float*)d_in[18];
    const float* W_out    = (const float*)d_in[19];

    float* ws    = (float*)d_ws;
    float* puk   = ws;                 // 512*1024
    float* xact  = ws + 524288;        // 512*2048
    float* sz    = ws + 1572864;       // 512*2048
    float* dtemb = ws + 2621440;       // 512*64
    float* xdbl  = ws + 2654208;       // 512*192
    float* rowd  = ws + 2752512;       // 512
    float* ybuf  = ws + 2753024;       // 512*2048

    float* hk = (float*)d_out;         // 512*1024
    float* ew = hk + 524288;           // 512*4 expert_weights
    float* em = hk + 526336;           // 512*4 expert_mask

    // 1) router / embeddings / KAN diff  -> dtemb, ew, em
    small_ops_kernel<<<512, 64, 0, stream>>>(tkc, tkp, rW, rb, freq, phase,
                                             kan_W, dtemb, ew, em);

    // 2) puk = uk @ W_in.T + b_in        (512x1024x1024)
    {
        GemmArgs g = {};
        g.A0 = uk;   g.lda0 = 1024; g.A1 = uk;   g.lda1 = 1024;
        g.B0 = W_in; g.ldb0 = 1024; g.B1 = W_in; g.ldb1 = 1024;
        g.out = puk; g.ldc = 1024;  g.K = 1024;  g.kSplit = 1024;
        g.mode = 1;  g.bias = b_in;
        gemm_f16_wmma<<<dim3(16, 16), 32, 0, stream>>>(g);
    }

    // 3) xz = puk @ W_inproj.T ; x = silu(conv(x)), z = silu(z)   (512x4096x1024)
    {
        GemmArgs g = {};
        g.A0 = puk;      g.lda0 = 1024; g.A1 = puk;      g.lda1 = 1024;
        g.B0 = W_inproj; g.ldb0 = 1024; g.B1 = W_inproj; g.ldb1 = 1024;
        g.out = xact; g.ldc = 2048; g.out2 = sz;
        g.K = 1024; g.kSplit = 1024;
        g.mode = 2; g.aux0 = conv_w; g.aux1 = conv_b;
        gemm_f16_wmma<<<dim3(64, 16), 32, 0, stream>>>(g);
    }

    // 4) xdbl = xact @ W_x.T             (512x192x2048)
    {
        GemmArgs g = {};
        g.A0 = xact; g.lda0 = 2048; g.A1 = xact; g.lda1 = 2048;
        g.B0 = W_x;  g.ldb0 = 2048; g.B1 = W_x;  g.ldb1 = 2048;
        g.out = xdbl; g.ldc = 192; g.K = 2048; g.kSplit = 2048; g.mode = 0;
        gemm_f16_wmma<<<dim3(3, 16), 32, 0, stream>>>(g);
    }

    // 5) rowdot[b] = B_row . C_row
    rowdot_kernel<<<2, 256, 0, stream>>>(xdbl, rowd);

    // 6) dt = softplus(dt_r@W_dt.T + dtemb@W_delta.T + dt_bias)
    //    ybuf = (dt*rowdot + D_skip) * xact * sz       (512x2048x128, split K)
    {
        GemmArgs g = {};
        g.A0 = xdbl;  g.lda0 = 192; g.A1 = dtemb;   g.lda1 = 64;
        g.B0 = W_dt;  g.ldb0 = 64;  g.B1 = W_delta; g.ldb1 = 64;
        g.out = ybuf; g.ldc = 2048; g.K = 128; g.kSplit = 64;
        g.mode = 4; g.bias = dt_bias;
        g.aux0 = rowd; g.aux1 = D_skip; g.aux2 = xact; g.aux3 = sz;
        gemm_f16_wmma<<<dim3(32, 16), 32, 0, stream>>>(g);
    }

    // 7) hk = ybuf @ W_out.T             (512x1024x2048)
    {
        GemmArgs g = {};
        g.A0 = ybuf;  g.lda0 = 2048; g.A1 = ybuf;  g.lda1 = 2048;
        g.B0 = W_out; g.ldb0 = 2048; g.B1 = W_out; g.ldb1 = 2048;
        g.out = hk; g.ldc = 1024; g.K = 2048; g.kSplit = 2048; g.mode = 0;
        gemm_f16_wmma<<<dim3(16, 16), 32, 0, stream>>>(g);
    }
}
